// PokerTransformerV1_44710609551624
// MI455X (gfx1250) — matrix operand order repaired and checked by reading, not compiled, not run on hardware
//
#include <hip/hip_runtime.h>
#include <hip/hip_bf16.h>

typedef __bf16 bf16;
typedef __attribute__((ext_vector_type(16))) __bf16 v16bf;
typedef __attribute__((ext_vector_type(8)))  __bf16 v8bf;
typedef __attribute__((ext_vector_type(8)))  float  v8f;

#define B_   8
#define TN   128
#define CM   2048
#define DM   1024
#define NH   16
#define DH   64
#define MT   (CM + TN)   // 2176

// ---------- small helpers ----------
static __device__ inline v8f zero8f() {
  v8f z;
#pragma unroll
  for (int i = 0; i < 8; ++i) z[i] = 0.0f;
  return z;
}

static __device__ inline v16bf zero16bf() {
  v16bf z;
#pragma unroll
  for (int i = 0; i < 16; ++i) z[i] = (bf16)0.0f;
  return z;
}

static __device__ inline v16bf cat8(v8bf a, v8bf b) {
  v16bf r;
#pragma unroll
  for (int i = 0; i < 8; ++i) { r[i] = a[i]; r[8 + i] = b[i]; }
  return r;
}

static __device__ inline v16bf cvt16(const float* __restrict__ p) {
  v16bf r;
#pragma unroll
  for (int i = 0; i < 16; ++i) r[i] = (bf16)p[i];
  return r;
}

static __device__ inline v8f wmma_bf16(v16bf a, v16bf b, v8f c) {
  return __builtin_amdgcn_wmma_f32_16x16x32_bf16(false, a, false, b, (short)0, c,
                                                 false, false);
}

// reduce across the 16 lanes of this lane's half-wave (wave32)
static __device__ inline float halfmax(float x) {
  x = fmaxf(x, __shfl_xor(x, 1, 32));
  x = fmaxf(x, __shfl_xor(x, 2, 32));
  x = fmaxf(x, __shfl_xor(x, 4, 32));
  x = fmaxf(x, __shfl_xor(x, 8, 32));
  return x;
}
static __device__ inline float halfsum(float x) {
  x += __shfl_xor(x, 1, 32);
  x += __shfl_xor(x, 2, 32);
  x += __shfl_xor(x, 4, 32);
  x += __shfl_xor(x, 8, 32);
  return x;
}

// ---------- kernel 0: fp32 -> bf16 conversion ----------
__global__ __launch_bounds__(256) void cvt_kernel(const float* __restrict__ src,
                                                  bf16* __restrict__ dst, int n4) {
  int i = blockIdx.x * blockDim.x + threadIdx.x;
  if (i < n4) {
    float4 v = ((const float4*)src)[i];
    bf16* d = dst + i * 4;
    d[0] = (bf16)v.x; d[1] = (bf16)v.y; d[2] = (bf16)v.z; d[3] = (bf16)v.w;
  }
}

// ---------- kernel 1: QKV projection + bias + RoPE + valid mask ----------
// one wave computes a 16(M) x 64(head) tile of one of {q,k,v}
__global__ __launch_bounds__(256) void qkv_rope_kernel(
    const bf16* __restrict__ xb,
    const bf16* __restrict__ wq, const bf16* __restrict__ wk,
    const bf16* __restrict__ wv,
    const float* __restrict__ bq, const float* __restrict__ bk,
    const float* __restrict__ bv,
    const float* __restrict__ inv_freq,
    const int* __restrict__ plen_, const int* __restrict__ nlen_,
    bf16* __restrict__ qb, bf16* __restrict__ kb, bf16* __restrict__ vb) {
  const int wave = threadIdx.x >> 5;
  const int lane = threadIdx.x & 31;
  const int tile = blockIdx.x * 8 + wave;             // 0..3071
  if (tile >= 3 * 64 * NH) return;
  const int mat = tile / (64 * NH);
  const int rem = tile % (64 * NH);
  const int mt = rem >> 4;                            // M tile 0..63
  const int ht = rem & 15;                            // head
  const int m0 = mt * 16;
  const int b  = m0 / TN;
  const int t0 = m0 % TN;
  const int nl = lane & 15;
  const int hi = lane >> 4;

  const bf16*  W    = (mat == 0) ? wq : (mat == 1) ? wk : wv;
  const float* bias = (mat == 0) ? bq : (mat == 1) ? bk : bv;
  bf16*        out  = (mat == 0) ? qb : (mat == 1) ? kb : vb;

  v8f acc[4];
#pragma unroll
  for (int j = 0; j < 4; ++j) acc[j] = zero8f();

  const bf16* arow = xb + (size_t)(m0 + nl) * DM;     // A row m = lane%16
  for (int k0 = 0; k0 < DM; k0 += 32) {
    v16bf a = *(const v16bf*)(arow + k0 + 16 * hi);
#pragma unroll
    for (int j = 0; j < 4; ++j) {
      const bf16* wrow = W + (size_t)(ht * DH + j * 16 + nl) * DM + k0 + 16 * hi;
      v16bf bf = *(const v16bf*)wrow;
      acc[j] = wmma_bf16(a, bf, acc[j]);
    }
  }

  const int plen = plen_[b];
  const int nlen = nlen_[b];
  float biasj[4];
#pragma unroll
  for (int j = 0; j < 4; ++j) biasj[j] = bias[ht * DH + j * 16 + nl];
  const float invf0 = inv_freq[nl];
  const float invf1 = inv_freq[16 + nl];

  float y[4][8];
  if (mat < 2) {  // q or k: RoPE
#pragma unroll
    for (int r = 0; r < 8; ++r) {
      int t = t0 + r + 8 * hi;
      float valid = (t < nlen) ? 1.0f : 0.0f;
      float pos = (float)(plen + t);
      float a0 = pos * invf0, a1 = pos * invf1;
      float c0 = __cosf(a0), s0 = __sinf(a0);
      float c1 = __cosf(a1), s1 = __sinf(a1);
      float x0 = acc[0][r] + biasj[0];
      float x1 = acc[1][r] + biasj[1];
      float x2 = acc[2][r] + biasj[2];
      float x3 = acc[3][r] + biasj[3];
      y[0][r] = (x0 * c0 - x2 * s0) * valid;  // d in [0,16)
      y[1][r] = (x1 * c1 - x3 * s1) * valid;  // d in [16,32)
      y[2][r] = (x2 * c0 + x0 * s0) * valid;  // d in [32,48)
      y[3][r] = (x3 * c1 + x1 * s1) * valid;  // d in [48,64)
    }
  } else {  // v: bias + mask only
#pragma unroll
    for (int r = 0; r < 8; ++r) {
      int t = t0 + r + 8 * hi;
      float valid = (t < nlen) ? 1.0f : 0.0f;
#pragma unroll
      for (int j = 0; j < 4; ++j) y[j][r] = (acc[j][r] + biasj[j]) * valid;
    }
  }

  // store [B, H, TN, DH] bf16
#pragma unroll
  for (int r = 0; r < 8; ++r) {
    int t = t0 + r + 8 * hi;
    bf16* orow = out + (((size_t)b * NH + ht) * TN + t) * DH;
#pragma unroll
    for (int j = 0; j < 4; ++j) orow[j * 16 + nl] = (bf16)y[j][r];
  }
}

// ---------- kernel 2: flash attention over past + new KV ----------
// one wave per (b, h, 16-query tile); 32 keys per iteration
__global__ __launch_bounds__(256) void attn_kernel(
    const bf16* __restrict__ qb, const bf16* __restrict__ kb,
    const bf16* __restrict__ vb,
    const float* __restrict__ past_k, const float* __restrict__ past_v,
    const int* __restrict__ plen_, const int* __restrict__ nlen_,
    bf16* __restrict__ ctxb) {
  __shared__ bf16 ldsP[8][16 * 32];    // per-wave probability tile (16 q x 32 k)
  __shared__ bf16 ldsVt[8][64 * 32];   // per-wave transposed V block (64 d x 32 k)

  const int wave = threadIdx.x >> 5;
  const int lane = threadIdx.x & 31;
  const int tile = blockIdx.x * 8 + wave;             // 0..1023
  const int b  = tile >> 7;
  const int h  = (tile >> 3) & 15;
  const int mt = tile & 7;
  const int m0 = mt * 16;
  const int nl = lane & 15;
  const int hi = lane >> 4;

  const int plen = plen_[b];
  const int tlen = plen + nlen_[b];

  // Q A-fragments (K-dim = DH = 64 -> two 32-wide steps), loaded once
  const bf16* qrow = qb + (((size_t)b * NH + h) * TN + m0 + nl) * DH;
  v16bf qa0 = *(const v16bf*)(qrow + 16 * hi);
  v16bf qa1 = *(const v16bf*)(qrow + 32 + 16 * hi);

  const float* pkb = past_k + ((size_t)b * NH + h) * CM * DH;
  const float* pvb = past_v + ((size_t)b * NH + h) * CM * DH;
  const bf16*  knb = kb + ((size_t)b * NH + h) * TN * DH;
  const bf16*  vnb = vb + ((size_t)b * NH + h) * TN * DH;

  v8f Oacc[4];
#pragma unroll
  for (int s = 0; s < 4; ++s) Oacc[s] = zero8f();
  float Mrow[8], Lrow[8];
#pragma unroll
  for (int r = 0; r < 8; ++r) { Mrow[r] = -3.0e38f; Lrow[r] = 0.0f; }

  bf16* Pw = &ldsP[wave][0];
  bf16* Vw = &ldsVt[wave][0];

  for (int j0 = 0; j0 < MT; j0 += 32) {
    if (j0 >= tlen) break;

    // prefetch next block's KV rows (speculative, cache-line granular)
    if (j0 + 32 + lane < plen) {
      __builtin_prefetch(pkb + (size_t)(j0 + 32 + lane) * DH, 0, 0);
      __builtin_prefetch(pvb + (size_t)(j0 + 32 + lane) * DH, 0, 0);
    }

    // ---- K B-fragments: two 16-key subtiles x two 32-d k-steps
    v16bf kfr[2][2];
#pragma unroll
    for (int tg = 0; tg < 2; ++tg) {
      int key = j0 + tg * 16 + nl;
      if (key < plen) {
        const float* kr = pkb + (size_t)key * DH;
        kfr[tg][0] = cvt16(kr + 16 * hi);
        kfr[tg][1] = cvt16(kr + 32 + 16 * hi);
      } else if (key < tlen) {
        const bf16* kr = knb + (size_t)(key - plen) * DH;
        kfr[tg][0] = *(const v16bf*)(kr + 16 * hi);
        kfr[tg][1] = *(const v16bf*)(kr + 32 + 16 * hi);
      } else {
        kfr[tg][0] = zero16bf();
        kfr[tg][1] = zero16bf();
      }
    }

    // ---- scores S = Q K^T  (two 16x16 accumulators)
    v8f S0 = zero8f(), S1 = zero8f();
    S0 = wmma_bf16(qa0, kfr[0][0], S0);
    S0 = wmma_bf16(qa1, kfr[0][1], S0);
    S1 = wmma_bf16(qa0, kfr[1][0], S1);
    S1 = wmma_bf16(qa1, kfr[1][1], S1);

    // ---- online softmax (row stats via half-wave reductions)
    const bool v0 = (j0 + nl) < tlen;
    const bool v1 = (j0 + 16 + nl) < tlen;
#pragma unroll
    for (int r = 0; r < 8; ++r) {
      float s0 = v0 ? S0[r] * 0.125f : -3.0e38f;
      float s1 = v1 ? S1[r] * 0.125f : -3.0e38f;
      float mnew = fmaxf(Mrow[r], halfmax(fmaxf(s0, s1)));
      float alpha = __expf(Mrow[r] - mnew);
      float p0 = __expf(s0 - mnew);
      float p1 = __expf(s1 - mnew);
      Mrow[r] = mnew;
      Lrow[r] = Lrow[r] * alpha + halfsum(p0 + p1);
      Oacc[0][r] *= alpha; Oacc[1][r] *= alpha;
      Oacc[2][r] *= alpha; Oacc[3][r] *= alpha;
      Pw[(r + 8 * hi) * 32 + nl] = (bf16)p0;
      Pw[(r + 8 * hi) * 32 + 16 + nl] = (bf16)p1;
    }

    // ---- stage V block transposed into LDS: Vt[d][kk], this lane owns key kk=lane
    {
      int key = j0 + lane;
      const float* pvr = pvb + (size_t)key * DH;
      int knidx = key - plen;
      if (knidx < 0 || knidx >= TN) knidx = 0;
      const bf16* vnr = vnb + (size_t)knidx * DH;
#pragma unroll
      for (int d0 = 0; d0 < 64; d0 += 8) {
#pragma unroll
        for (int i = 0; i < 8; ++i) {
          float f;
          if (key < plen)      f = pvr[d0 + i];
          else if (key < tlen) f = (float)vnr[d0 + i];
          else                 f = 0.0f;
          Vw[(d0 + i) * 32 + lane] = (bf16)f;
        }
      }
    }

    // same-wave DS ops are in-order; fence the compiler + wait DS completion
    asm volatile("s_wait_dscnt 0" ::: "memory");

    // ---- P as A-fragment (16 x 32) from LDS
    const bf16* pr = Pw + nl * 32;
    v16bf pa = *(const v16bf*)(pr + 16 * hi);

    // ---- ctx += P @ V  (4 n-subtiles covering d = 0..63)
#pragma unroll
    for (int sub = 0; sub < 4; ++sub) {
      v16bf vfr = *(const v16bf*)(Vw + (sub * 16 + nl) * 32 + 16 * hi);
      Oacc[sub] = wmma_bf16(pa, vfr, Oacc[sub]);
    }
  }

  // ---- normalize and store ctx as bf16 [B*TN, DM] with col = h*64 + d
#pragma unroll
  for (int r = 0; r < 8; ++r) {
    float rl = 1.0f / Lrow[r];
    int t = m0 + r + 8 * hi;
    bf16* crow = ctxb + (size_t)(b * TN + t) * DM + h * DH;
#pragma unroll
    for (int sub = 0; sub < 4; ++sub)
      crow[sub * 16 + nl] = (bf16)(Oacc[sub][r] * rl);
  }
}

// ---------- kernel 3: output projection + bias + valid mask ----------
__global__ __launch_bounds__(256) void oproj_kernel(
    const bf16* __restrict__ ctxb, const bf16* __restrict__ wob,
    const float* __restrict__ bo, const int* __restrict__ nlen_,
    float* __restrict__ out) {
  const int wave = threadIdx.x >> 5;
  const int lane = threadIdx.x & 31;
  const int tile = blockIdx.x * 8 + wave;   // 0..4095
  const int mt = tile >> 6;
  const int nt = tile & 63;
  const int m0 = mt * 16, n0 = nt * 16;
  const int b = m0 / TN, t0 = m0 % TN;
  const int nl = lane & 15;
  const int hi = lane >> 4;

  v8f acc = zero8f();
  const bf16* arow = ctxb + (size_t)(m0 + nl) * DM;
  const bf16* brow = wob + (size_t)(n0 + nl) * DM;
  for (int k0 = 0; k0 < DM; k0 += 32) {
    v16bf a = *(const v16bf*)(arow + k0 + 16 * hi);
    v16bf bb = *(const v16bf*)(brow + k0 + 16 * hi);
    acc = wmma_bf16(a, bb, acc);
  }

  const int nlen = nlen_[b];
  const float bias = bo[n0 + nl];
#pragma unroll
  for (int r = 0; r < 8; ++r) {
    int t = t0 + r + 8 * hi;
    float yv = (t < nlen) ? (acc[r] + bias) : 0.0f;
    out[(size_t)(b * TN + t) * DM + n0 + nl] = yv;
  }
}

// ---------- host launcher ----------
extern "C" void kernel_launch(void* const* d_in, const int* in_sizes, int n_in,
                              void* d_out, int out_size, void* d_ws, size_t ws_size,
                              hipStream_t stream) {
  const float* x_new    = (const float*)d_in[0];
  const float* past_k   = (const float*)d_in[1];
  const float* past_v   = (const float*)d_in[2];
  const float* Wq       = (const float*)d_in[3];
  const float* bq       = (const float*)d_in[4];
  const float* Wk       = (const float*)d_in[5];
  const float* bk       = (const float*)d_in[6];
  const float* Wv       = (const float*)d_in[7];
  const float* bv       = (const float*)d_in[8];
  const float* Wo       = (const float*)d_in[9];
  const float* bo       = (const float*)d_in[10];
  const float* inv_freq = (const float*)d_in[11];
  const int*   plen     = (const int*)d_in[12];
  const int*   nlen     = (const int*)d_in[13];

  const size_t SZ = (size_t)DM * DM;          // 1,048,576 elements (== B*TN*DM)
  char* ws = (char*)d_ws;
  bf16* xb   = (bf16*)(ws + 0 * SZ * 2);
  bf16* wqb  = (bf16*)(ws + 1 * SZ * 2);
  bf16* wkb  = (bf16*)(ws + 2 * SZ * 2);
  bf16* wvb  = (bf16*)(ws + 3 * SZ * 2);
  bf16* wob  = (bf16*)(ws + 4 * SZ * 2);
  bf16* qbuf = (bf16*)(ws + 5 * SZ * 2);
  bf16* kbuf = (bf16*)(ws + 6 * SZ * 2);
  bf16* vbuf = (bf16*)(ws + 7 * SZ * 2);
  bf16* ctxb = (bf16*)(ws + 8 * SZ * 2);      // 18 MB total

  const int n4 = (int)(SZ / 4);
  const int cblocks = (n4 + 255) / 256;
  cvt_kernel<<<cblocks, 256, 0, stream>>>(x_new, xb, n4);
  cvt_kernel<<<cblocks, 256, 0, stream>>>(Wq, wqb, n4);
  cvt_kernel<<<cblocks, 256, 0, stream>>>(Wk, wkb, n4);
  cvt_kernel<<<cblocks, 256, 0, stream>>>(Wv, wvb, n4);
  cvt_kernel<<<cblocks, 256, 0, stream>>>(Wo, wob, n4);

  qkv_rope_kernel<<<384, 256, 0, stream>>>(xb, wqb, wkb, wvb, bq, bk, bv,
                                           inv_freq, plen, nlen,
                                           qbuf, kbuf, vbuf);

  attn_kernel<<<128, 256, 0, stream>>>(qbuf, kbuf, vbuf, past_k, past_v,
                                       plen, nlen, ctxb);

  oproj_kernel<<<512, 256, 0, stream>>>(ctxb, wob, bo, nlen, (float*)d_out);
}
